// CNF_69217692942625
// MI455X (gfx1250) — compile-verified, matrix operand order checked by reading, not run on hardware
//
#include <hip/hip_runtime.h>

// ---------------------------------------------------------------------------
// CNF log-likelihood on gfx1250 (MI455X).
//
// Per vf-evaluation, propagate an augmented matrix per sample
//   A = [ J (32 x din) ; y (1 x din) ]  through each ConcatSquash layer:
//   Z = A @ W1^T ; forward row -> h = z*gate + (b1*gate + W3*t), y' = softplus(h)
//   J rows -> scaled by s = gate * sigmoid(h)  (gate only for last layer)
//   divergence = trace of final 32x32 Jacobian.
// GEMMs on v_wmma_f32_16x16x32_bf16; weights pre-swizzled to fragment-ready
// bf16; gate/c folded per (layer,time) since the Tsit5 grid is fixed.
// Nonlinearities use raw v_exp/v_log/v_rcp so they co-execute with WMMA.
// nt-loops unrolled x2 to give the scheduler two independent WMMA chains and
// overlap B-fragment loads with the XDL pipe.
// ---------------------------------------------------------------------------

typedef __bf16 bf16_t;
typedef bf16_t v16bf __attribute__((ext_vector_type(16)));
typedef float  v8f   __attribute__((ext_vector_type(8)));
typedef unsigned int u32;
typedef u32    v4u   __attribute__((ext_vector_type(4)));

#define PITCH   264          // padded LDS row pitch (bf16 elems) to spread banks
#define NSAMP   4            // samples per workgroup
#define NTILE_A 9            // 8 Jacobian row-tiles + 1 forward tile
#define FWD_T   8
#define HSTEP   (-0.1f)
#define WBLK_E  147456       // bf16 elems of swizzled W per block
#define GC_LSTR 30720        // float stride per layer in gate/c table (60*512)
#define LOG2E   1.4426950408889634f
#define LN2     0.6931471805599453f

__constant__ float ACF[6][5] = {
  {0.f,0.f,0.f,0.f,0.f},
  {0.161f,0.f,0.f,0.f,0.f},
  {-0.008480655492356989f, 0.335480655492357f, 0.f,0.f,0.f},
  {2.8971530571054935f, -6.359448489975075f, 4.3622954328695815f, 0.f,0.f},
  {5.325864828439257f, -11.748883564062828f, 7.4955393428898365f, -0.09249506636175525f, 0.f},
  {5.86145544294642f, -12.92096931784711f, 8.159367898576159f, -0.071584973281401f, -0.028269050394068383f}};
__constant__ float BCF[6] = {0.09646076681806523f, 0.01f, 0.4798896504144996f,
                             1.379008574103742f, -3.290069515436081f, 2.324710524099774f};
__constant__ float CCF[6] = {0.f, 0.161f, 0.327f, 0.9f, 0.9800255409045097f, 1.f};

struct WPtrs {
  const float* W1[8]; const float* W2[8]; const float* W3[8];
  const float* b1[8]; const float* b2[8];
};

union FragU { v4u u[2]; v16bf v; };

__device__ __forceinline__ unsigned short f2bf(float f) {
  u32 u = __float_as_uint(f);
  u32 r = u + 0x7FFFu + ((u >> 16) & 1u);      // RNE bf16
  return (unsigned short)(r >> 16);
}
// Fast hardware transcendentals: single v_exp_f32 / v_log_f32 / v_rcp_f32,
// co-executable with the WMMA (XDL) pipe.
__device__ __forceinline__ float fast_sigmoid(float x) {
  return __builtin_amdgcn_rcpf(1.f + __builtin_amdgcn_exp2f(-x * LOG2E));
}
__device__ __forceinline__ float fast_softplus(float x) {
  float ax = __builtin_fabsf(x);
  float e  = __builtin_amdgcn_exp2f(-ax * LOG2E);
  float l  = __builtin_amdgcn_logf(1.f + e) * LN2;   // log1p(exp(-|x|))
  return fmaxf(x, 0.f) + l;
}

// A fragment (16x32 bf16, lane = row M, two 8-wide K runs) from LDS row-major.
__device__ __forceinline__ v16bf load_afrag(const unsigned short* A, int tile, int kt, int lane) {
  int m  = lane & 15;
  int kh = (lane & 16) ? 8 : 0;
  const unsigned short* p = A + (tile * 16 + m) * PITCH + kt * 32 + kh;
  FragU u;
  u.u[0] = *(const v4u*)(p);
  u.u[1] = *(const v4u*)(p + 16);
  return u.v;
}
// B fragment (32x16 bf16) from pre-swizzled global: lane-contiguous 32B.
__device__ __forceinline__ v16bf load_bfrag(const unsigned short* wf, int frag, int lane) {
  const unsigned short* p = wf + frag * 512 + lane * 16;
  FragU u;
  u.u[0] = *(const v4u*)(p);
  u.u[1] = *(const v4u*)(p + 8);
  return u.v;
}
__device__ __forceinline__ v8f bf_wmma(v16bf a, v16bf b, v8f c) {
  return __builtin_amdgcn_wmma_f32_16x16x32_bf16(false, a, false, b, (short)0, c, false, false);
}

// One ConcatSquash layer over the augmented LDS matrix.
template<int KT, int NTILES, bool LAST>
__device__ __forceinline__ void layer_pass(
    const unsigned short* Ain, unsigned short* Aout,
    const unsigned short* wf, const float* gct,
    float* Ssc, float* Ff, float* Dpart) {
  const int lane = threadIdx.x & 31;
  const int wave = threadIdx.x >> 5;
  __builtin_prefetch(wf, 0, 3);   // global_prefetch_b8: pull W frags toward WGP

  if (wave == 0) {                // forward tile first: produces h -> S scales
    v16bf afr[KT];
#pragma unroll
    for (int kt = 0; kt < KT; ++kt) afr[kt] = load_afrag(Ain, FWD_T, kt, lane);
#pragma unroll 2
    for (int nt = 0; nt < NTILES; ++nt) {
      v8f acc = {};
#pragma unroll
      for (int kt = 0; kt < KT; ++kt)
        acc = bf_wmma(afr[kt], load_bfrag(wf, kt * NTILES + nt, lane), acc);
      int col = nt * 16 + (lane & 15);
      if (!LAST) {
        if (lane < 16) {
          float g = gct[col], cc = gct[256 + col];
#pragma unroll
          for (int v = 0; v < 8; ++v) {
            unsigned short ov = 0;
            if (v < NSAMP) {
              float h = acc[v] * g + cc;
              Ssc[v * 256 + col] = g * fast_sigmoid(h);
              ov = f2bf(fast_softplus(h));
            }
            Aout[(FWD_T * 16 + v) * PITCH + col] = ov;
          }
        } else {
#pragma unroll
          for (int v = 0; v < 8; ++v)
            Aout[(FWD_T * 16 + 8 + v) * PITCH + col] = 0;
        }
      } else {
        if (lane < 16) {
          float g = gct[col], cc = gct[256 + col];
#pragma unroll
          for (int v = 0; v < NSAMP; ++v) Ff[v * 32 + col] = acc[v] * g + cc;
        }
      }
    }
  }
  __syncthreads();                // S (or gate-only last layer) ready

  {                               // Jacobian tiles: one 16-row tile per wave
    const int tile = wave;
    const int sample = tile >> 1;
    v16bf afr[KT];
#pragma unroll
    for (int kt = 0; kt < KT; ++kt) afr[kt] = load_afrag(Ain, tile, kt, lane);
    const int ntb = LAST ? (tile & 1) : 0;       // last layer: only diagonal tile
    const int nte = LAST ? (ntb + 1) : NTILES;
#pragma unroll 2
    for (int nt = ntb; nt < nte; ++nt) {
      v8f acc = {};
#pragma unroll
      for (int kt = 0; kt < KT; ++kt)
        acc = bf_wmma(afr[kt], load_bfrag(wf, kt * NTILES + nt, lane), acc);
      int col = nt * 16 + (lane & 15);
      if (!LAST) {
        float s = Ssc[sample * 256 + col];
        int mb = (lane & 16) ? 8 : 0;
#pragma unroll
        for (int v = 0; v < 8; ++v)
          Aout[(tile * 16 + mb + v) * PITCH + col] = f2bf(acc[v] * s);
      } else {
        // diagonal of final Jacobian: C layout -> lanes 0-7 (M=v=lane) and
        // lanes 24-31 (M=v+8=lane-16) carry a diag element when col==M.
        int vsel = (lane < 8) ? lane : ((lane >= 24) ? (lane - 24) : -1);
        float av = 0.f;
#pragma unroll
        for (int i = 0; i < 8; ++i) av = (vsel == i) ? acc[i] : av;
        Dpart[tile * 32 + lane] = (vsel >= 0) ? gct[col] * av : 0.f;
      }
    }
  }
}

// --- pre-pass 1: gate/c tables over the fixed Tsit5 time grid -------------
__global__ void cnf_gatec(WPtrs wp, float* gcs) {
  int b = blockIdx.x;
  int li = b / 60, tidx = b % 60;
  int l = li & 3, n = threadIdx.x;
  int dout = (l == 3) ? 32 : 256;
  int step = tidx / 6, st = tidx % 6;
  float tt = 1.0f - 0.1f * (float)step - 0.1f * CCF[st];
  float* dst = gcs + (size_t)(li * 60 + tidx) * 512;
  float g = 0.f, c = 0.f;
  if (n < dout) {
    float w2 = wp.W2[li][n], bb2 = wp.b2[li][n];
    float bb1 = wp.b1[li][n], w3 = wp.W3[li][n];
    g = 1.f / (1.f + expf(-(w2 * tt + bb2)));   // pre-pass: precise is fine
    c = bb1 * g + w3 * tt;
  }
  dst[n] = g;
  dst[256 + n] = c;
}

// --- pre-pass 2: W1 -> bf16 fragment-ready swizzle -------------------------
__global__ void cnf_swz(WPtrs wp, unsigned short* wdst) {
  int e = blockIdx.x * 256 + threadIdx.x;
  const int total = 2 * WBLK_E;
  if (e >= total) return;
  int blk = e / WBLK_E, r = e % WBLK_E;
  int l, off, Kd, ntiles;
  if (r < 8192)        { l = 0; off = 0;      Kd = 32;  ntiles = 16; }
  else if (r < 73728)  { l = 1; off = 8192;   Kd = 256; ntiles = 16; }
  else if (r < 139264) { l = 2; off = 73728;  Kd = 256; ntiles = 16; }
  else                 { l = 3; off = 139264; Kd = 256; ntiles = 2;  }
  int f = r - off;
  int frag = f >> 9, idx = f & 511;
  int lane = idx >> 4, slot = idx & 15;
  int v = slot >> 1, p = slot & 1;
  int kt = frag / ntiles, nt = frag % ntiles;
  int n = nt * 16 + (lane & 15);
  int klc = ((v < 4) ? (v * 2 + p) : (16 + (v - 4) * 2 + p)) + ((lane >= 16) ? 8 : 0);
  int k = kt * 32 + klc;
  int li = blk * 4 + l;
  wdst[e] = f2bf(wp.W1[li][n * Kd + k]);
}

// --- main persistent CNF kernel -------------------------------------------
__global__ __launch_bounds__(256)
void cnf_main(const float* __restrict__ Yin, const unsigned short* __restrict__ wsW,
              const float* __restrict__ wsG, float* __restrict__ out) {
  __shared__ __align__(16) unsigned short As0[NTILE_A * 16 * PITCH];
  __shared__ __align__(16) unsigned short As1[NTILE_A * 16 * PITCH];
  __shared__ float Ssc[NSAMP * 256];
  __shared__ float Ff[NSAMP * 32];
  __shared__ float Dpart[8 * 32];
  __shared__ float kyS[6][NSAMP][32];
  __shared__ float klS[6][NSAMP];
  __shared__ float ybase[NSAMP][32];
  __shared__ float ysd[NSAMP][32];
  __shared__ float lpS[NSAMP];

  const int tid = threadIdx.x;
  if (tid < 128) {
    int sm = tid >> 5, d = tid & 31;
    float yv = Yin[((size_t)blockIdx.x * NSAMP + sm) * 32 + d];
    ybase[sm][d] = yv;
    ysd[sm][d] = yv;
  }
  if (tid < NSAMP) lpS[tid] = 0.f;
  __syncthreads();

  for (int bi = 0; bi < 2; ++bi) {
    const int blk = 1 - bi;                                 // reversed(params)
    const unsigned short* wb = wsW + (size_t)blk * WBLK_E;
    const float* gb = wsG + (size_t)blk * 4 * 60 * 512;
    for (int step = 0; step < 10; ++step) {
      for (int st = 0; st < 6; ++st) {
        const int tidx = step * 6 + st;
        const float* g0 = gb + (size_t)tidx * 512;

        // Build layer-0 augmented input: identity Jacobian rows + ys rows.
        for (int e = tid; e < NTILE_A * 16 * 32; e += 256) {
          int t = e >> 9, m = (e >> 5) & 15, k = e & 31;
          unsigned short val;
          if (t < 8) { int cot = ((t & 1) << 4) + m; val = (cot == k) ? (unsigned short)0x3F80u : 0; }
          else       { val = (m < NSAMP) ? f2bf(ysd[m][k]) : (unsigned short)0; }
          As0[(t * 16 + m) * PITCH + k] = val;
        }
        __syncthreads();

        layer_pass<1, 16, false>(As0, As1, wb,          g0,               Ssc, Ff, Dpart);
        __syncthreads();
        layer_pass<8, 16, false>(As1, As0, wb + 8192,   g0 + GC_LSTR,     Ssc, Ff, Dpart);
        __syncthreads();
        layer_pass<8, 16, false>(As0, As1, wb + 73728,  g0 + 2 * GC_LSTR, Ssc, Ff, Dpart);
        __syncthreads();
        layer_pass<8, 2,  true >(As1, As0, wb + 139264, g0 + 3 * GC_LSTR, Ssc, Ff, Dpart);
        __syncthreads();

        // Tsit5 stage bookkeeping (deterministic LDS reductions).
        if (tid < 128) {
          int sm = tid >> 5, d = tid & 31;
          kyS[st][sm][d] = Ff[sm * 32 + d];
        }
        if (tid < NSAMP) {
          float dv = 0.f;
          for (int i = 0; i < 64; ++i) dv += Dpart[tid * 64 + i];
          klS[st][tid] = dv;
        }
        __syncthreads();
        if (tid < 128) {
          int sm = tid >> 5, d = tid & 31;
          if (st < 5) {
            float a = ybase[sm][d];
            for (int j = 0; j <= st; ++j) a += HSTEP * ACF[st + 1][j] * kyS[j][sm][d];
            ysd[sm][d] = a;
          } else {
            float a = ybase[sm][d];
#pragma unroll
            for (int j = 0; j < 6; ++j) a += HSTEP * BCF[j] * kyS[j][sm][d];
            ybase[sm][d] = a;
            ysd[sm][d] = a;
          }
        }
        if (tid < NSAMP && st == 5) {
          float a = 0.f;
#pragma unroll
          for (int j = 0; j < 6; ++j) a += BCF[j] * klS[j][tid];
          lpS[tid] += HSTEP * a;
        }
        __syncthreads();
      }
    }
  }

  if (tid < NSAMP) {
    float s2 = 0.f;
    for (int d = 0; d < 32; ++d) { float v = ybase[tid][d]; s2 += v * v; }
    out[(size_t)blockIdx.x * NSAMP + tid] =
        lpS[tid] - 0.5f * (32.f * 1.8378770664093453f + s2);
  }
}

extern "C" void kernel_launch(void* const* d_in, const int* in_sizes, int n_in,
                              void* d_out, int out_size, void* d_ws, size_t ws_size,
                              hipStream_t stream) {
  (void)in_sizes; (void)n_in; (void)out_size; (void)ws_size;
  WPtrs wp;
  for (int li = 0; li < 8; ++li) {
    int ib = 1 + li * 5;            // jax dict key order: W1, W2, W3, b1, b2
    wp.W1[li] = (const float*)d_in[ib + 0];
    wp.W2[li] = (const float*)d_in[ib + 1];
    wp.W3[li] = (const float*)d_in[ib + 2];
    wp.b1[li] = (const float*)d_in[ib + 3];
    wp.b2[li] = (const float*)d_in[ib + 4];
  }
  unsigned short* wsWp = (unsigned short*)d_ws;                 // 589,824 B
  float* wsGp = (float*)((char*)d_ws + (size_t)2 * WBLK_E * 2); // 983,040 B

  cnf_gatec<<<480, 256, 0, stream>>>(wp, wsGp);
  cnf_swz<<<(2 * WBLK_E + 255) / 256, 256, 0, stream>>>(wp, wsWp);
  cnf_main<<<4096 / NSAMP, 256, 0, stream>>>((const float*)d_in[0], wsWp, wsGp,
                                             (float*)d_out);
}
